// DecoderWithAttention_62130996904637
// MI455X (gfx1250) — compile-verified, matrix operand order checked by reading, not run on hardware
//
#include <hip/hip_runtime.h>
#include <hip/hip_bf16.h>
#include <math.h>

// Problem constants (match reference)
#define B_    256
#define S_    512
#define ENC_  256
#define DEC_  256
#define ATTN_ 128
#define H_    48
#define NG_   1024   // 4*DEC
#define KG_   512    // ENC + DEC (inp column handled as rank-1 epilogue)

typedef __attribute__((ext_vector_type(16))) __bf16          v16bf;
typedef __attribute__((ext_vector_type(8)))  float           v8f;
typedef __attribute__((ext_vector_type(8)))  unsigned short  v8u16;
typedef __attribute__((ext_vector_type(4)))  unsigned short  v4u16;

union BF16x16 { v16bf v; v8u16 h[2]; };
union BF16x4  { v4u16 q; unsigned short u[4]; };

__device__ __forceinline__ unsigned short f2bf(float x) {
  union { float f; unsigned u; } c; c.f = x;
  unsigned r = c.u + 0x7FFFu + ((c.u >> 16) & 1u);   // round-to-nearest-even
  return (unsigned short)(r >> 16);
}
__device__ __forceinline__ float bf2f(unsigned short h) {
  union { unsigned u; float f; } c; c.u = ((unsigned)h) << 16;
  return c.f;
}
__device__ __forceinline__ float sigm(float x) { return 1.0f / (1.0f + __expf(-x)); }

// ---- WMMA fragment loads (CDNA5 16-bit layout: lane L: group lg=L/16,
// elements i<8 -> K = lg*8+i ; i>=8 -> K = 16 + lg*8 + (i-8)) ----
__device__ __forceinline__ v16bf load_afrag(const unsigned short* base, int row0,
                                            int stride, int k0) {
  int lane = threadIdx.x & 31;
  int lg = lane >> 4;
  int m  = lane & 15;
  const unsigned short* p = base + (size_t)(row0 + m) * stride + k0 + lg * 8;
  BF16x16 f;
  f.h[0] = *reinterpret_cast<const v8u16*>(p);        // K = lg*8 + [0..7]
  f.h[1] = *reinterpret_cast<const v8u16*>(p + 16);   // K = 16 + lg*8 + [0..7]
  return f.v;
}
__device__ __forceinline__ v16bf load_bfrag(const unsigned short* packed, int frag) {
  int lane = threadIdx.x & 31;
  const unsigned short* p = packed + ((size_t)frag * 32 + lane) * 16;
  BF16x16 f;
  f.h[0] = *reinterpret_cast<const v8u16*>(p);
  f.h[1] = *reinterpret_cast<const v8u16*>(p + 8);
  return f.v;
}

// ---------------- Preprocessing ----------------

// h0/c0 -> state buffers; h0 also into bf16 activation row [b][256+d]
__global__ void k_init(const float* __restrict__ h0, const float* __restrict__ c0,
                       float* __restrict__ hbuf, float* __restrict__ cbuf,
                       unsigned short* __restrict__ a_in) {
  int idx = blockIdx.x * 256 + threadIdx.x;          // 65536
  hbuf[idx] = h0[idx];
  cbuf[idx] = c0[idx];
  int b = idx >> 8, d = idx & 255;
  a_in[(size_t)b * KG_ + ENC_ + d] = f2bf(h0[idx]);
}

// Pack gate weight B-matrix: Wg[n][k] = (k<256 ? W_ih[n][1+k] : W_hh[n][k-256])
// into fragment-major bf16: frag = nTile*16 + kstep  (64 x 16 frags)
__global__ void k_pack_wg(const float* __restrict__ W_ih, const float* __restrict__ W_hh,
                          unsigned short* __restrict__ packWg) {
  int frag = blockIdx.x;                              // 1024
  int lane = threadIdx.x;                             // 32
  int nTile = frag >> 4, kstep = frag & 15;
  int lg = lane >> 4, n = lane & 15;
  int nG = nTile * 16 + n;
  for (int i = 0; i < 16; ++i) {
    int kk = kstep * 32 + (i < 8 ? lg * 8 + i : 16 + lg * 8 + (i - 8));
    float v = (kk < ENC_) ? W_ih[(size_t)nG * 257 + 1 + kk]
                          : W_hh[(size_t)nG * 256 + (kk - ENC_)];
    packWg[((size_t)frag * 32 + lane) * 16 + i] = f2bf(v);
  }
}

// Pack W_enc (ENC x ATTN, K x N) fragments: frag = nTile*8 + kstep (8 x 8)
__global__ void k_pack_wenc(const float* __restrict__ W_enc,
                            unsigned short* __restrict__ packWenc) {
  int frag = blockIdx.x;                              // 64
  int lane = threadIdx.x;                             // 32
  int nTile = frag >> 3, kstep = frag & 7;
  int lg = lane >> 4, n = lane & 15;
  for (int i = 0; i < 16; ++i) {
    int kk = kstep * 32 + (i < 8 ? lg * 8 + i : 16 + lg * 8 + (i - 8));
    packWenc[((size_t)frag * 32 + lane) * 16 + i] = f2bf(W_enc[(size_t)kk * ATTN_ + nTile * 16 + n]);
  }
}

// enc_outputs f32 -> bf16 (streamed once; per-step reads then cost half BW)
__global__ void k_cvt_enc(const float* __restrict__ enc, unsigned short* __restrict__ encB) {
  size_t i0 = ((size_t)blockIdx.x * 256 + threadIdx.x) * 8;   // grid 16384
  for (int i = 0; i < 8; ++i) encB[i0 + i] = f2bf(enc[i0 + i]);
}

// enc_proj = enc @ W_enc : M = B*S = 131072, N = 128, K = 256; bf16 WMMA
__global__ void k_encproj(const unsigned short* __restrict__ encB,
                          const unsigned short* __restrict__ packWenc,
                          unsigned short* __restrict__ eprojB) {
  int wv = threadIdx.x >> 5;
  int tile = blockIdx.x * 8 + wv;                     // 65536 tiles
  int mTile = tile >> 3, nTile = tile & 7;
  int lane = threadIdx.x & 31, lg = lane >> 4;
  v8f acc = {0.f, 0.f, 0.f, 0.f, 0.f, 0.f, 0.f, 0.f};
  for (int ks = 0; ks < 8; ++ks) {
    v16bf a = load_afrag(encB, mTile * 16, ENC_, ks * 32);
    v16bf b = load_bfrag(packWenc, nTile * 8 + ks);
    acc = __builtin_amdgcn_wmma_f32_16x16x32_bf16(false, a, false, b,
                                                  (short)0, acc, false, false);
  }
  int col = nTile * 16 + (lane & 15);
  for (int r = 0; r < 8; ++r) {
    int row = mTile * 16 + r + lg * 8;                // C/D layout: M = r + 8*lg
    eprojB[(size_t)row * ATTN_ + col] = f2bf(acc[r]);
  }
}

// ---------------- Per-timestep kernels ----------------

// One block per batch row: dec_proj -> tanh scores -> softmax -> context
__global__ void k_attn(const unsigned short* __restrict__ encB,
                       const unsigned short* __restrict__ eprojB,
                       const float* __restrict__ W_dec, const float* __restrict__ vvec,
                       const float* __restrict__ hbuf,
                       unsigned short* __restrict__ a_in,
                       float* __restrict__ d_out, int t) {
  __shared__ float hs[DEC_];
  __shared__ float dproj[ATTN_];
  __shared__ float sc[S_];
  __shared__ float red[256];
  int b = blockIdx.x, tid = threadIdx.x;

  hs[tid] = hbuf[(size_t)b * DEC_ + tid];
  __syncthreads();

  if (tid < ATTN_) {                                  // dec_proj = h @ W_dec
    float acc = 0.f;
    for (int d = 0; d < DEC_; ++d) acc += hs[d] * W_dec[(size_t)d * ATTN_ + tid];
    dproj[tid] = acc;
  }
  __syncthreads();

  // scores: one wave per s-value, lanes split ATTN
  int wv = tid >> 5, lane = tid & 31;
  const unsigned short* epb = eprojB + (size_t)b * S_ * ATTN_;
  for (int s = wv; s < S_; s += 8) {
    BF16x4 q; q.q = *reinterpret_cast<const v4u16*>(epb + (size_t)s * ATTN_ + lane * 4);
    float p = 0.f;
    for (int j = 0; j < 4; ++j) {
      int a = lane * 4 + j;
      p += vvec[a] * tanhf(bf2f(q.u[j]) + dproj[a]);
    }
    for (int off = 16; off >= 1; off >>= 1) p += __shfl_xor(p, off, 32);
    if (lane == 0) sc[s] = p;
  }
  __syncthreads();

  // block softmax over S=512 (2 values per thread)
  red[tid] = fmaxf(sc[tid], sc[tid + 256]);
  __syncthreads();
  for (int off = 128; off >= 1; off >>= 1) {
    if (tid < off) red[tid] = fmaxf(red[tid], red[tid + off]);
    __syncthreads();
  }
  float mx = red[0];
  __syncthreads();
  float e0 = __expf(sc[tid] - mx), e1 = __expf(sc[tid + 256] - mx);
  sc[tid] = e0; sc[tid + 256] = e1; red[tid] = e0 + e1;
  __syncthreads();
  for (int off = 128; off >= 1; off >>= 1) {
    if (tid < off) red[tid] += red[tid + off];
    __syncthreads();
  }
  float inv = 1.0f / red[0];
  sc[tid] *= inv; sc[tid + 256] *= inv;
  float* attn_out = d_out + (size_t)B_ * H_ + ((size_t)b * H_ + t) * S_;
  attn_out[tid] = sc[tid]; attn_out[tid + 256] = sc[tid + 256];
  __syncthreads();

  // context[e] = sum_s w[s] * enc[b,s,e]  (coalesced bf16 stream)
  const unsigned short* ep = encB + (size_t)b * S_ * ENC_ + tid;
  float ctx = 0.f;
  for (int s = 0; s < S_; ++s) {
    if (s + 8 < S_) __builtin_prefetch(ep + (size_t)(s + 8) * ENC_, 0, 1);
    ctx += sc[s] * bf2f(ep[(size_t)s * ENC_]);
  }
  a_in[(size_t)b * KG_ + tid] = f2bf(ctx);
}

// gates = [ctx,h] @ Wg^T (+ biases + inp * W_ih[:,0]) : M=256, N=1024, K=512 WMMA
__global__ void k_gates(const unsigned short* __restrict__ a_in,
                        const unsigned short* __restrict__ packWg,
                        const float* __restrict__ b_ih, const float* __restrict__ b_hh,
                        const float* __restrict__ W_ih, const float* __restrict__ d_out,
                        float* __restrict__ gates, int t) {
  int wv = threadIdx.x >> 5;
  int tile = blockIdx.x * 8 + wv;                     // 1024 tiles = 16 M x 64 N
  int mTile = tile >> 6, nTile = tile & 63;
  int lane = threadIdx.x & 31, lg = lane >> 4;
  v8f acc = {0.f, 0.f, 0.f, 0.f, 0.f, 0.f, 0.f, 0.f};
  for (int ks = 0; ks < 16; ++ks) {
    v16bf a = load_afrag(a_in, mTile * 16, KG_, ks * 32);
    v16bf b = load_bfrag(packWg, nTile * 16 + ks);
    acc = __builtin_amdgcn_wmma_f32_16x16x32_bf16(false, a, false, b,
                                                  (short)0, acc, false, false);
  }
  int n = nTile * 16 + (lane & 15);
  float bias = b_ih[n] + b_hh[n];
  float w0 = W_ih[(size_t)n * 257];                   // inp column (rank-1 term)
  for (int r = 0; r < 8; ++r) {
    int b_row = mTile * 16 + r + lg * 8;
    float inp = (t == 0) ? 0.f : d_out[(size_t)b_row * H_ + (t - 1)];
    gates[(size_t)b_row * NG_ + n] = acc[r] + bias + inp * w0;
  }
}

// LSTM pointwise update + output projection (one block per batch row)
__global__ void k_lstm(const float* __restrict__ gates, float* __restrict__ cbuf,
                       float* __restrict__ hbuf, unsigned short* __restrict__ a_in,
                       const float* __restrict__ W_out, const float* __restrict__ b_out,
                       float* __restrict__ d_out, int t) {
  __shared__ float red[256];
  int b = blockIdx.x, d = threadIdx.x;
  const float* g = gates + (size_t)b * NG_;
  float gi = g[d], gf = g[DEC_ + d], gg = g[2 * DEC_ + d], go = g[3 * DEC_ + d];
  float c = cbuf[(size_t)b * DEC_ + d];
  float cn = sigm(gf) * c + sigm(gi) * tanhf(gg);
  float hn = sigm(go) * tanhf(cn);
  cbuf[(size_t)b * DEC_ + d] = cn;
  hbuf[(size_t)b * DEC_ + d] = hn;
  a_in[(size_t)b * KG_ + ENC_ + d] = f2bf(hn);
  red[d] = hn * W_out[d];
  __syncthreads();
  for (int off = 128; off >= 1; off >>= 1) {
    if (d < off) red[d] += red[d + off];
    __syncthreads();
  }
  if (d == 0) d_out[(size_t)b * H_ + t] = red[0] + b_out[0];
}

// ---------------- Host launcher ----------------
extern "C" void kernel_launch(void* const* d_in, const int* in_sizes, int n_in,
                              void* d_out, int out_size, void* d_ws, size_t ws_size,
                              hipStream_t stream) {
  (void)in_sizes; (void)n_in; (void)out_size; (void)ws_size;
  const float* enc   = (const float*)d_in[0];
  const float* h0    = (const float*)d_in[1];
  const float* c0    = (const float*)d_in[2];
  const float* W_enc = (const float*)d_in[3];
  const float* W_dec = (const float*)d_in[4];
  const float* vvec  = (const float*)d_in[5];
  const float* W_ih  = (const float*)d_in[6];
  const float* W_hh  = (const float*)d_in[7];
  const float* b_ih  = (const float*)d_in[8];
  const float* b_hh  = (const float*)d_in[9];
  const float* W_out = (const float*)d_in[10];
  const float* b_out = (const float*)d_in[11];
  float* out = (float*)d_out;

  // Workspace carve-up (256B aligned)
  char* ws = (char*)d_ws;
  size_t off = 0;
  auto carve = [&](size_t bytes) { char* p = ws + off; off = (off + bytes + 255) & ~(size_t)255; return p; };
  unsigned short* encB     = (unsigned short*)carve((size_t)B_ * S_ * ENC_ * 2);   // 64 MB
  unsigned short* eprojB   = (unsigned short*)carve((size_t)B_ * S_ * ATTN_ * 2);  // 32 MB
  unsigned short* a_in     = (unsigned short*)carve((size_t)B_ * KG_ * 2);
  unsigned short* packWg   = (unsigned short*)carve((size_t)NG_ * KG_ * 2);
  unsigned short* packWenc = (unsigned short*)carve((size_t)64 * 32 * 16 * 2);
  float*          gates    = (float*)carve((size_t)B_ * NG_ * 4);
  float*          cbuf     = (float*)carve((size_t)B_ * DEC_ * 4);
  float*          hbuf     = (float*)carve((size_t)B_ * DEC_ * 4);

  // One-time preprocessing
  k_init<<<256, 256, 0, stream>>>(h0, c0, hbuf, cbuf, a_in);
  k_pack_wg<<<1024, 32, 0, stream>>>(W_ih, W_hh, packWg);
  k_pack_wenc<<<64, 32, 0, stream>>>(W_enc, packWenc);
  k_cvt_enc<<<16384, 256, 0, stream>>>(enc, encB);
  k_encproj<<<8192, 256, 0, stream>>>(encB, packWenc, eprojB);

  // Recurrence: 48 dependent steps, 3 kernels each
  for (int t = 0; t < H_; ++t) {
    k_attn<<<B_, 256, 0, stream>>>(encB, eprojB, W_dec, vvec, hbuf, a_in, out, t);
    k_gates<<<128, 256, 0, stream>>>(a_in, packWg, b_ih, b_hh, W_ih, out, gates, t);
    k_lstm<<<B_, 256, 0, stream>>>(gates, cbuf, hbuf, a_in, W_out, b_out, out, t);
  }
}
